// PT_Attention_34291018891955
// MI455X (gfx1250) — matrix operand order repaired and checked by reading, not compiled, not run on hardware
//
#include <hip/hip_runtime.h>

// Problem constants (match reference)
#define B_  2
#define N_  1024
#define L_  2048
#define C_  256
#define H_  8
#define HD_ 32
#define LCHUNK 32
#define SCALE_ 0.17677669529663687f   // 32^-0.5

typedef __attribute__((ext_vector_type(16))) __bf16 v16bf;
typedef __attribute__((ext_vector_type(8)))  __bf16 v8bf;
typedef __attribute__((ext_vector_type(8)))  float  v8f;

static __device__ __forceinline__ __bf16 f2bf(float f) {
  unsigned int u = __builtin_bit_cast(unsigned int, f);
  unsigned int r = (u + 0x7FFFu + ((u >> 16) & 1u)) >> 16;   // round-to-nearest-even
  unsigned short s = (unsigned short)r;
  return __builtin_bit_cast(__bf16, s);
}

// ---------------- conversion kernels ----------------
__global__ void cvt_f32_bf16(const float* __restrict__ src, __bf16* __restrict__ dst, int n) {
  int i = blockIdx.x * blockDim.x + threadIdx.x;
  if (i < n) dst[i] = f2bf(src[i]);
}

// W is [in=C][out=C]; produce Wt[out][in] bf16 so B-operand reads are contiguous in K.
__global__ void cvtT_f32_bf16(const float* __restrict__ src, __bf16* __restrict__ dst) {
  int idx = blockIdx.x * blockDim.x + threadIdx.x;   // over C*C
  int i = idx >> 8;          // input dim
  int o = idx & 255;         // output dim
  dst[(size_t)o * C_ + i] = f2bf(src[idx]);
}

// ---------------- bf16 GEMM with A-fragment reuse: C = A[M][256] @ Wt^T ----------------
// Wt is [Nout=256][256] (pre-transposed weight). 4 waves/block; each wave computes a
// 16x64 strip: one A fragment is reused across 4 B fragments / 4 accumulators per K-step.
__global__ void gemm64_bf16(const __bf16* __restrict__ A, const __bf16* __restrict__ Bt,
                            __bf16* __restrict__ Cb, float* __restrict__ Cf,
                            const float* __restrict__ bias)
{
  const int lane  = threadIdx.x & 31;
  const int wv    = threadIdx.x >> 5;        // 0..3
  const int m0    = blockIdx.x * 16;
  const int n0    = wv * 64;                 // 64-col strip per wave; 4 waves cover 256

  const int arow  = lane & 15;
  const int kb_   = (lane < 16) ? 0 : 16;    // A-layout K base (contiguous: k = 16h + s)
  const int bn    = lane & 15;               // B-layout N index
  const int koffB = (lane < 16) ? 0 : 16;    // B-layout K group

  const __bf16* __restrict__ arp  = A  + (size_t)(m0 + arow) * C_;
  const __bf16* __restrict__ brp0 = Bt + (size_t)(n0 +  0 + bn) * C_ + koffB;
  const __bf16* __restrict__ brp1 = Bt + (size_t)(n0 + 16 + bn) * C_ + koffB;
  const __bf16* __restrict__ brp2 = Bt + (size_t)(n0 + 32 + bn) * C_ + koffB;
  const __bf16* __restrict__ brp3 = Bt + (size_t)(n0 + 48 + bn) * C_ + koffB;

  v8f acc0 = {}, acc1 = {}, acc2 = {}, acc3 = {};
#pragma unroll
  for (int k0 = 0; k0 < C_; k0 += 32) {
    v8bf lo = *(const v8bf*)(arp + k0 + kb_);
    v8bf hi = *(const v8bf*)(arp + k0 + 8 + kb_);
    v16bf a;
#pragma unroll
    for (int i = 0; i < 8; ++i) { a[i] = lo[i]; a[i + 8] = hi[i]; }
    v16bf b0 = *(const v16bf*)(brp0 + k0);
    v16bf b1 = *(const v16bf*)(brp1 + k0);
    v16bf b2 = *(const v16bf*)(brp2 + k0);
    v16bf b3 = *(const v16bf*)(brp3 + k0);
    acc0 = __builtin_amdgcn_wmma_f32_16x16x32_bf16(false, a, false, b0, (short)0, acc0, false, false);
    acc1 = __builtin_amdgcn_wmma_f32_16x16x32_bf16(false, a, false, b1, (short)0, acc1, false, false);
    acc2 = __builtin_amdgcn_wmma_f32_16x16x32_bf16(false, a, false, b2, (short)0, acc2, false, false);
    acc3 = __builtin_amdgcn_wmma_f32_16x16x32_bf16(false, a, false, b3, (short)0, acc3, false, false);
  }

  const int rb = (lane < 16) ? 0 : 8;        // D-layout row base
  if (Cf) {
#pragma unroll
    for (int j = 0; j < 4; ++j) {
      const v8f acc = (j == 0) ? acc0 : (j == 1) ? acc1 : (j == 2) ? acc2 : acc3;
      const int nc = n0 + j * 16 + bn;
      float bv = bias ? bias[nc] : 0.0f;
#pragma unroll
      for (int r = 0; r < 8; ++r)
        Cf[(size_t)(m0 + rb + r) * C_ + nc] = acc[r] + bv;
    }
  } else {
#pragma unroll
    for (int j = 0; j < 4; ++j) {
      const v8f acc = (j == 0) ? acc0 : (j == 1) ? acc1 : (j == 2) ? acc2 : acc3;
      const int nc = n0 + j * 16 + bn;
#pragma unroll
      for (int r = 0; r < 8; ++r)
        Cb[(size_t)(m0 + rb + r) * C_ + nc] = f2bf(acc[r]);
    }
  }
}

// ---------------- fused attention: scores + head-mix MLP mask + online softmax + P@V --------
// One block per (batch, 16-row q tile); 8 waves, wave w = head w.
// V chunks are staged into LDS with async global->LDS copies (ASYNCcnt), bypassing VGPRs.
__global__ void attn_kernel(const __bf16* __restrict__ Qp, const __bf16* __restrict__ Kp,
                            const __bf16* __restrict__ Vp, __bf16* __restrict__ Ob,
                            const float* __restrict__ W1, const float* __restrict__ b1,
                            const float* __restrict__ W2, const float* __restrict__ b2,
                            float* __restrict__ maskOut)
{
  __shared__ float  sc[H_][16][LCHUNK];     // scaled raw scores (all heads)
  __shared__ __bf16 vsh[LCHUNK][C_];        // staged V chunk (all heads)
  __shared__ float  mprev[H_][16], lsum[H_][16], mrow[H_][16], srow[H_][16];
  __shared__ float  wsh[81];                // W1(64) b1(8) W2(8) b2(1)

  const int tid  = threadIdx.x;
  const int lane = tid & 31;
  const int wv   = tid >> 5;                // head index
  const int bb   = blockIdx.x >> 6;         // batch
  const int n0   = (blockIdx.x & 63) * 16;  // q-row tile

  const size_t qbase  = (size_t)bb * N_ * C_;
  const size_t kvbase = (size_t)bb * L_ * C_;

  if (tid < 64)       wsh[tid] = W1[tid];
  else if (tid < 72)  wsh[tid] = b1[tid - 64];
  else if (tid < 80)  wsh[tid] = W2[tid - 72];
  else if (tid == 80) wsh[tid] = b2[0];
  if (tid < 128) { mprev[tid >> 4][tid & 15] = -3.0e38f; lsum[tid >> 4][tid & 15] = 0.0f; }
  __syncthreads();

  // ---- Q fragment for this head (A operand, 16x32), loaded once ----
  const int arow  = lane & 15;
  const int kb_   = (lane < 16) ? 0 : 16;
  const int rb    = (lane < 16) ? 0 : 8;
  const int bl    = lane & 15;              // B-operand N index
  const int koffB = (lane < 16) ? 0 : 16;   // B-operand K group
  v16bf qa;
  {
    const __bf16* qrp = Qp + qbase + (size_t)(n0 + arow) * C_ + wv * HD_;
    v8bf lo = *(const v8bf*)(qrp + kb_);
    v8bf hi = *(const v8bf*)(qrp + 8 + kb_);
#pragma unroll
    for (int i = 0; i < 8; ++i) { qa[i] = lo[i]; qa[i + 8] = hi[i]; }
  }

  // per-thread staging slice: 32 bf16 (64B) of the V chunk, copied async to LDS
  const int vr  = tid >> 3;
  const int vcb = (tid & 7) * 32;
  const unsigned ldsV = (unsigned)(size_t)&vsh[vr][vcb];   // low 32 bits = LDS byte offset

  v8f o0 = {}, o1 = {};                     // O accumulator 16x32 (two 16x16 tiles)

  for (int l0 = 0; l0 < L_; l0 += LCHUNK) {
    // 1. async-stage V chunk [32][256] into LDS (2 x b128 per thread, no VGPR round-trip)
    {
      unsigned long long gaV =
          (unsigned long long)(size_t)(Vp + kvbase + (size_t)(l0 + vr) * C_ + vcb);
      asm volatile(
        "global_load_async_to_lds_b128 %0, %1, off\n\t"
        "global_load_async_to_lds_b128 %0, %1, off offset:16"
        :: "v"(ldsV), "v"(gaV)
        : "memory");
    }
    // 2. scores for this head: Q(16x32) @ K^T(32x32l) -> two 16x16 WMMAs
    v8f s0 = {}, s1 = {};
    {
      const __bf16* kp0 = Kp + kvbase + (size_t)(l0 + bl) * C_ + wv * HD_ + koffB;
      const __bf16* kp1 = Kp + kvbase + (size_t)(l0 + 16 + bl) * C_ + wv * HD_ + koffB;
      v16bf kf0 = *(const v16bf*)kp0;
      v16bf kf1 = *(const v16bf*)kp1;
      if (l0 + LCHUNK < L_)
        __builtin_prefetch((const void*)(kp0 + (size_t)LCHUNK * C_), 0, 1);  // global_prefetch_b8
      s0 = __builtin_amdgcn_wmma_f32_16x16x32_bf16(false, qa, false, kf0, (short)0, s0, false, false);
      s1 = __builtin_amdgcn_wmma_f32_16x16x32_bf16(false, qa, false, kf1, (short)0, s1, false, false);
    }
#pragma unroll
    for (int r = 0; r < 8; ++r) {
      sc[wv][rb + r][bl]      = s0[r] * SCALE_;
      sc[wv][rb + r][16 + bl] = s1[r] * SCALE_;
    }
    // drain this wave's async V copies, then make them visible block-wide at the barrier
    asm volatile("s_wait_asynccnt 0x0" ::: "memory");
    __syncthreads();

    // 3. head-mix MLP mask (needs all 8 heads): 512 (n,l) pairs / 256 threads
    for (int p = tid; p < 16 * LCHUNK; p += 256) {
      int rn = p >> 5, rl = p & 31;
      float hv[8];
#pragma unroll
      for (int h = 0; h < 8; ++h) hv[h] = sc[h][rn][rl];
      float acc2 = wsh[80];
#pragma unroll
      for (int j = 0; j < 8; ++j) {
        float f = wsh[64 + j];
#pragma unroll
        for (int h = 0; h < 8; ++h) f += hv[h] * wsh[h * 8 + j];
        acc2 += fmaxf(f, 0.0f) * wsh[72 + j];
      }
      maskOut[((size_t)bb * N_ + n0 + rn) * L_ + l0 + rl] = fmaxf(acc2, 0.0f);
    }

    // 4. online-softmax row stats (lanes 0-15, one row each)
    if (lane < 16) {
      int rn = lane;
      float mp = mprev[wv][rn];
      float mx = mp;
#pragma unroll
      for (int j = 0; j < LCHUNK; ++j) mx = fmaxf(mx, sc[wv][rn][j]);
      float sscale = __expf(mp - mx);
      float rs = 0.0f;
#pragma unroll
      for (int j = 0; j < LCHUNK; ++j) rs += __expf(sc[wv][rn][j] - mx);
      lsum[wv][rn]  = lsum[wv][rn] * sscale + rs;
      mprev[wv][rn] = mx;
      mrow[wv][rn]  = mx;
      srow[wv][rn]  = sscale;
    }
    __syncthreads();

    // 5. rescale O accumulators
#pragma unroll
    for (int r = 0; r < 8; ++r) {
      float scl = srow[wv][rb + r];
      o0[r] *= scl; o1[r] *= scl;
    }

    // 6. build P in A-fragment layout (LDS read doubles as D->A transpose)
    v16bf pa;
    {
      float mrv = mrow[wv][arow];
#pragma unroll
      for (int i = 0; i < 16; ++i) {
        int kk = i + kb_;
        pa[i] = f2bf(__expf(sc[wv][arow][kk] - mrv));
      }
    }
    // 7. V fragments (B operand, K=l) from LDS, then P(16x32) @ V(32x32d)
    {
      int dsel = lane & 15;
      int lofs = (lane < 16) ? 0 : 16;
      v16bf vf0, vf1;
#pragma unroll
      for (int i = 0; i < 16; ++i) {
        vf0[i] = vsh[lofs + i][wv * HD_ + dsel];
        vf1[i] = vsh[lofs + i][wv * HD_ + 16 + dsel];
      }
      o0 = __builtin_amdgcn_wmma_f32_16x16x32_bf16(false, pa, false, vf0, (short)0, o0, false, false);
      o1 = __builtin_amdgcn_wmma_f32_16x16x32_bf16(false, pa, false, vf1, (short)0, o1, false, false);
    }
    __syncthreads();   // protect sc/vsh before next iteration
  }

  // epilogue: normalize and store O (bf16) at this head's column block
#pragma unroll
  for (int r = 0; r < 8; ++r) {
    float inv = 1.0f / lsum[wv][rb + r];
    size_t orow = qbase + (size_t)(n0 + rb + r) * C_ + wv * HD_;
    Ob[orow + (lane & 15)]      = f2bf(o0[r] * inv);
    Ob[orow + 16 + (lane & 15)] = f2bf(o1[r] * inv);
  }
}

// ---------------- host-side launcher ----------------
extern "C" void kernel_launch(void* const* d_in, const int* in_sizes, int n_in,
                              void* d_out, int out_size, void* d_ws, size_t ws_size,
                              hipStream_t stream) {
  const float* query = (const float*)d_in[0];
  const float* key   = (const float*)d_in[1];
  const float* value = (const float*)d_in[2];
  const float* Wq    = (const float*)d_in[3];
  const float* Wk    = (const float*)d_in[4];
  const float* Wv    = (const float*)d_in[5];
  const float* Wp    = (const float*)d_in[6];
  const float* bp    = (const float*)d_in[7];
  const float* W1    = (const float*)d_in[8];
  const float* b1    = (const float*)d_in[9];
  const float* W2    = (const float*)d_in[10];
  const float* b2    = (const float*)d_in[11];

  const int nQ = B_ * N_ * C_;   // 524288
  const int nK = B_ * L_ * C_;   // 1048576
  const int nW = C_ * C_;        // 65536

  char* ws = (char*)d_ws;
  __bf16* qb  = (__bf16*)ws;                 ws += (size_t)nQ * 2;
  __bf16* kb  = (__bf16*)ws;                 ws += (size_t)nK * 2;
  __bf16* vb  = (__bf16*)ws;                 ws += (size_t)nK * 2;
  __bf16* WqT = (__bf16*)ws;                 ws += (size_t)nW * 2;
  __bf16* WkT = (__bf16*)ws;                 ws += (size_t)nW * 2;
  __bf16* WvT = (__bf16*)ws;                 ws += (size_t)nW * 2;
  __bf16* WpT = (__bf16*)ws;                 ws += (size_t)nW * 2;
  __bf16* Qp  = (__bf16*)ws;                 ws += (size_t)nQ * 2;
  __bf16* Kp  = (__bf16*)ws;                 ws += (size_t)nK * 2;
  __bf16* Vp  = (__bf16*)ws;                 ws += (size_t)nK * 2;
  __bf16* Obuf= (__bf16*)ws;                 ws += (size_t)nQ * 2;

  float* xOut    = (float*)d_out;            // [B,N,C]
  float* maskOut = (float*)d_out + nQ;       // [B,N,L,1]

  // bf16 conversions + weight transposes
  cvt_f32_bf16<<<(nQ + 255) / 256, 256, 0, stream>>>(query, qb, nQ);
  cvt_f32_bf16<<<(nK + 255) / 256, 256, 0, stream>>>(key,   kb, nK);
  cvt_f32_bf16<<<(nK + 255) / 256, 256, 0, stream>>>(value, vb, nK);
  cvtT_f32_bf16<<<nW / 256, 256, 0, stream>>>(Wq, WqT);
  cvtT_f32_bf16<<<nW / 256, 256, 0, stream>>>(Wk, WkT);
  cvtT_f32_bf16<<<nW / 256, 256, 0, stream>>>(Wv, WvT);
  cvtT_f32_bf16<<<nW / 256, 256, 0, stream>>>(Wp, WpT);

  // QKV projections (WMMA bf16, 16x64 strip per wave)
  gemm64_bf16<<<(B_ * N_) / 16, 128, 0, stream>>>(qb, WqT, Qp, nullptr, nullptr);
  gemm64_bf16<<<(B_ * L_) / 16, 128, 0, stream>>>(kb, WkT, Kp, nullptr, nullptr);
  gemm64_bf16<<<(B_ * L_) / 16, 128, 0, stream>>>(vb, WvT, Vp, nullptr, nullptr);

  // fused attention + mask
  attn_kernel<<<B_ * (N_ / 16), 256, 0, stream>>>(Qp, Kp, Vp, Obuf, W1, b1, W2, b2, maskOut);

  // output projection with bias -> f32 result
  gemm64_bf16<<<(B_ * N_) / 16, 128, 0, stream>>>(Obuf, WpT, nullptr, xOut, bp);
}
